// WiSeGTN_34282428957141
// MI455X (gfx1250) — matrix operand order repaired and checked by reading, not compiled, not run on hardware
//
#include <hip/hip_runtime.h>
#include <stdint.h>

// ---------------------------------------------------------------------------
// WiSe-GTN for MI455X (gfx1250): all GEMMs via v_wmma_f32_16x16x32_bf16.
// B=2, H=8, L=256, C=512, heads=8, d=64, K(top)=4, 3 experts, 2 layers.
// GEMM waves compute 32x32 register tiles (2x2 wmma) to halve vmem per wmma.
// ---------------------------------------------------------------------------

#define Bn  2
#define Hn  8
#define Ln  256
#define Cn  512
#define Kt  4
#define NHn 8
#define DHn 64

typedef __attribute__((ext_vector_type(16))) __bf16 v16bf;
typedef __attribute__((ext_vector_type(8)))  float  v8f;

__device__ __forceinline__ unsigned short f32_bf16(float f) {
  union { float f; unsigned u; } x; x.f = f;
  unsigned r = x.u + 0x7FFFu + ((x.u >> 16) & 1u);   // round-to-nearest-even
  return (unsigned short)(r >> 16);
}

__device__ __forceinline__ v8f wmma_bf16(v16bf a, v16bf b, v8f c) {
  // (neg_a, A, neg_b, B, c_mod, C, reuse_a, reuse_b)
  return __builtin_amdgcn_wmma_f32_16x16x32_bf16(false, a, false, b, (short)0, c,
                                                 false, false);
}

// A fragment, 16x32 bf16, row-major source (ld in elements).
// Lanes 0-15 hold K=0..15, lanes 16-31 hold K=16..31 (matches load_b_nm),
// row M = lane&15. Pairs are K-contiguous -> dword loads (merge into b128).
__device__ __forceinline__ v16bf load_a_rm(const unsigned short* src, int ld, int kbase) {
  int lane = threadIdx.x & 31, m = lane & 15, half = lane >> 4;
  v16bf a;
  unsigned* ap = reinterpret_cast<unsigned*>(&a);
  const unsigned short* rp = src + (size_t)m * ld + kbase;
#pragma unroll
  for (int t = 0; t < 8; ++t) {
    int e0 = 2 * t;
    int k0 = e0 + 16 * half;
    ap[t] = *reinterpret_cast<const unsigned*>(rp + k0);
  }
  return a;
}

// B fragment, 32x16 bf16, from an n-major ("Bt") source: bt[n][k], ldt = K stride.
// Lanes 0-15: K=0..15, lanes 16-31: K=16..31, col N = lane&15.
__device__ __forceinline__ v16bf load_b_nm(const unsigned short* bt, int ldt, int kbase) {
  int lane = threadIdx.x & 31, n = lane & 15, half = lane >> 4;
  v16bf b;
  unsigned* bp = reinterpret_cast<unsigned*>(&b);
  const unsigned short* cp = bt + (size_t)n * ldt + kbase + 16 * half;
#pragma unroll
  for (int t = 0; t < 8; ++t)
    bp[t] = *reinterpret_cast<const unsigned*>(cp + 2 * t);
  return b;
}

// ---------------------------------------------------------------------------
// K1: top-4 + softmax of global_attn_map (B,H,H). 16 independent (b,j) rows.
// ---------------------------------------------------------------------------
__global__ void k_topk(const float* __restrict__ g, float* __restrict__ n_attn,
                       int* __restrict__ indx) {
  int t = threadIdx.x;
  if (t >= Bn * Hn) return;
  float v[Hn];
#pragma unroll
  for (int i = 0; i < Hn; ++i) v[i] = g[t * Hn + i];
  bool used[Hn] = {};
  float sel[Kt]; int si[Kt];
#pragma unroll
  for (int k = 0; k < Kt; ++k) {
    float best = -3.4e38f; int bi = 0;
#pragma unroll
    for (int i = 0; i < Hn; ++i)
      if (!used[i] && v[i] > best) { best = v[i]; bi = i; }
    used[bi] = true; sel[k] = best; si[k] = bi;
  }
  float mx = sel[0], s = 0.f;
#pragma unroll
  for (int k = 0; k < Kt; ++k) { sel[k] = __expf(sel[k] - mx); s += sel[k]; }
#pragma unroll
  for (int k = 0; k < Kt; ++k) { n_attn[t * Kt + k] = sel[k] / s; indx[t * Kt + k] = si[k]; }
}

// ---------------------------------------------------------------------------
// K2: weights f32 -> bf16, transposed to n-major Wt[mat][e][n][k].
// mat: 0=Wq 1=Wk 2=Wv 3=Wo. i covers (e,k,n) for all 4 mats.
// ---------------------------------------------------------------------------
__global__ void k_prepw(const float* __restrict__ wq, const float* __restrict__ wk,
                        const float* __restrict__ wv, const float* __restrict__ wo,
                        unsigned short* __restrict__ wt) {
  int i = blockIdx.x * 256 + threadIdx.x;               // 0 .. 3*C*C-1
  int e = i / (Cn * Cn);
  int r = i - e * Cn * Cn;
  int kk = r / Cn, nn = r - kk * Cn;
  size_t din = (size_t)nn * Cn + kk;
  wt[(size_t)(0 * 3 + e) * Cn * Cn + din] = f32_bf16(wq[i]);
  wt[(size_t)(1 * 3 + e) * Cn * Cn + din] = f32_bf16(wk[i]);
  wt[(size_t)(2 * 3 + e) * Cn * Cn + din] = f32_bf16(wv[i]);
  wt[(size_t)(3 * 3 + e) * Cn * Cn + din] = f32_bf16(wo[i]);
}

// ---------------------------------------------------------------------------
// K3: ctx = concat(history, current) -> f32 (residual) + bf16 (WMMA A).
// ---------------------------------------------------------------------------
__global__ void k_concat(const float* __restrict__ hist, const float* __restrict__ cur,
                         float* __restrict__ ctxf, unsigned short* __restrict__ ctxbf) {
  int i = blockIdx.x * 256 + threadIdx.x;               // < B*H*L*C
  int c = i & (Cn - 1);
  int l = (i >> 9) & (Ln - 1);
  int j = (i >> 17) & (Hn - 1);
  int b = i >> 20;
  float val = (j < Hn - 1)
            ? hist[(((size_t)b * (Hn - 1) + j) * Ln + l) * Cn + c]
            : cur[((size_t)b * Ln + l) * Cn + c];
  ctxf[i] = val;
  ctxbf[i] = f32_bf16(val);
}

// ---------------------------------------------------------------------------
// K4: QKV projections, all 3 experts. grid.z = e*3 + which (0=q,1=k,2=v).
// Block = 8 waves, each wave owns a 32x32 register tile (2 A x 2 B frags,
// 4 wmma per 32-K step) -> block tile 64x128.
// q,k stored row-major bf16; v stored transposed ([c][l]) for the P*V B operand.
// ---------------------------------------------------------------------------
__global__ __launch_bounds__(256) void k_qkv(
    const unsigned short* __restrict__ ctxbf, const unsigned short* __restrict__ wt,
    const float* __restrict__ bq, const float* __restrict__ bk, const float* __restrict__ bv,
    unsigned short* __restrict__ qb, unsigned short* __restrict__ kb,
    unsigned short* __restrict__ vtb) {
  int z = blockIdx.z, e = z / 3, which = z - e * 3;
  int w = threadIdx.x >> 5, wr = w >> 2, wc = w & 3;
  int row0 = blockIdx.y * 64 + wr * 32;                 // global (b,j,l) row
  int col0 = blockIdx.x * 128 + wc * 32;
  const unsigned short* A  = ctxbf + (size_t)row0 * Cn;
  const unsigned short* Bt = wt + (size_t)(which * 3 + e) * Cn * Cn + (size_t)col0 * Cn;

  v8f acc[2][2];
#pragma unroll
  for (int i = 0; i < 2; ++i)
#pragma unroll
    for (int jn = 0; jn < 2; ++jn)
#pragma unroll
      for (int q = 0; q < 8; ++q) acc[i][jn][q] = 0.f;

  for (int kk = 0; kk < Cn; kk += 32) {
    v16bf a0 = load_a_rm(A,                      Cn, kk);
    v16bf a1 = load_a_rm(A + (size_t)16 * Cn,    Cn, kk);
    v16bf b0 = load_b_nm(Bt,                     Cn, kk);
    v16bf b1 = load_b_nm(Bt + (size_t)16 * Cn,   Cn, kk);
    acc[0][0] = wmma_bf16(a0, b0, acc[0][0]);
    acc[0][1] = wmma_bf16(a0, b1, acc[0][1]);
    acc[1][0] = wmma_bf16(a1, b0, acc[1][0]);
    acc[1][1] = wmma_bf16(a1, b1, acc[1][1]);
  }

  const float* bias = (which == 0 ? bq : which == 1 ? bk : bv) + e * Cn;
  int lane = threadIdx.x & 31, n = lane & 15, half = lane >> 4;
  int bj = row0 >> 8, lbase = row0 & (Ln - 1);
#pragma unroll
  for (int jn = 0; jn < 2; ++jn) {
    float bs = bias[col0 + jn * 16 + n];
#pragma unroll
    for (int i = 0; i < 2; ++i)
#pragma unroll
      for (int v = 0; v < 8; ++v) {
        int lrow = lbase + i * 16 + v + 8 * half;
        int col  = col0 + jn * 16 + n;
        unsigned short val = f32_bf16(acc[i][jn][v] + bs);
        if (which == 2) {
          vtb[(size_t)(e * 16 + bj) * Cn * Ln + (size_t)col * Ln + lrow] = val;
        } else {
          unsigned short* dst = (which == 0) ? qb : kb;
          dst[((size_t)(e * 16 + bj) * Ln + lrow) * Cn + col] = val;
        }
      }
  }
}

// ---------------------------------------------------------------------------
// K5: attention for the SELECTED expert only (3x fewer FLOPs than reference).
// One block per (b,j,k-slot,head): full 256x256xd=64 attention.
// 8 waves * 16 query rows, two passes of 128 rows. S kept in 16 v8f accums,
// row max/sum via shfl_xor over the 16-lane column groups (wave32).
// P staged bf16 in a per-wave LDS slab, O = P*V via WMMA against v_t.
// ---------------------------------------------------------------------------
__global__ __launch_bounds__(256) void k_attn(
    const unsigned short* __restrict__ qb, const unsigned short* __restrict__ kb,
    const unsigned short* __restrict__ vtb, const int* __restrict__ indx,
    unsigned short* __restrict__ ao) {
  __shared__ unsigned short pbuf[8 * 16 * Ln];          // 64 KB, per-wave slabs

  int head = blockIdx.x;
  int slot = blockIdx.y;                                 // 0..63
  int kslot = slot & (Kt - 1), bj = slot >> 2;
  int j = bj & (Hn - 1), b = bj >> 3;
  int idx = indx[bj * Kt + kslot];
  int e = (idx == j) ? 0 : (idx > j ? 2 : 1);            // self / future / past

  const unsigned short* Qp = qb  + ((size_t)(e * 16 + bj) * Ln) * Cn + head * DHn;
  const unsigned short* Kp = kb  + ((size_t)(e * 16 + b * Hn + idx) * Ln) * Cn + head * DHn;
  const unsigned short* Vt = vtb + (size_t)(e * 16 + b * Hn + idx) * Cn * Ln
                                 + (size_t)head * DHn * Ln;

  int w = threadIdx.x >> 5, lane = threadIdx.x & 31, half = lane >> 4;
  unsigned short* pb = pbuf + w * 16 * Ln;
  size_t obase = ((size_t)bj * Kt + kslot) * Ln * Cn;

  for (int qo = 0; qo < Ln; qo += 128) {
    int r0 = qo + w * 16;
    v16bf qa0 = load_a_rm(Qp + (size_t)r0 * Cn, Cn, 0);
    v16bf qa1 = load_a_rm(Qp + (size_t)r0 * Cn, Cn, 32);

    v8f S[16];
#pragma unroll
    for (int ch = 0; ch < 16; ++ch) {
      v8f acc;
#pragma unroll
      for (int i = 0; i < 8; ++i) acc[i] = 0.f;
      const unsigned short* kr = Kp + (size_t)(ch * 16) * Cn;
      acc = wmma_bf16(qa0, load_b_nm(kr, Cn, 0),  acc);
      acc = wmma_bf16(qa1, load_b_nm(kr, Cn, 32), acc);
      S[ch] = acc;
    }

    // softmax: scale, row max, exp, row sum (rows live at element v, 16 lanes wide)
    float mx[8], sm[8];
#pragma unroll
    for (int v = 0; v < 8; ++v) mx[v] = -3.4e38f;
#pragma unroll
    for (int ch = 0; ch < 16; ++ch)
#pragma unroll
      for (int v = 0; v < 8; ++v) {
        S[ch][v] *= 0.125f;                              // 1/sqrt(64)
        mx[v] = fmaxf(mx[v], S[ch][v]);
      }
#pragma unroll
    for (int v = 0; v < 8; ++v) {
      mx[v] = fmaxf(mx[v], __shfl_xor(mx[v], 1, 32));
      mx[v] = fmaxf(mx[v], __shfl_xor(mx[v], 2, 32));
      mx[v] = fmaxf(mx[v], __shfl_xor(mx[v], 4, 32));
      mx[v] = fmaxf(mx[v], __shfl_xor(mx[v], 8, 32));
      sm[v] = 0.f;
    }
#pragma unroll
    for (int ch = 0; ch < 16; ++ch)
#pragma unroll
      for (int v = 0; v < 8; ++v) {
        float p = __expf(S[ch][v] - mx[v]);
        S[ch][v] = p; sm[v] += p;
      }
#pragma unroll
    for (int v = 0; v < 8; ++v) {
      sm[v] += __shfl_xor(sm[v], 1, 32);
      sm[v] += __shfl_xor(sm[v], 2, 32);
      sm[v] += __shfl_xor(sm[v], 4, 32);
      sm[v] += __shfl_xor(sm[v], 8, 32);
    }

    // P (16x256) -> per-wave LDS slab, row-major bf16
#pragma unroll
    for (int ch = 0; ch < 16; ++ch)
#pragma unroll
      for (int v = 0; v < 8; ++v) {
        int m = v + 8 * half, col = ch * 16 + (lane & 15);
        pb[m * Ln + col] = f32_bf16(S[ch][v]);
      }

    // O = P * V  (M=16, N=64, K=256), B operand from transposed V
    v8f O[4];
#pragma unroll
    for (int nc = 0; nc < 4; ++nc)
#pragma unroll
      for (int i = 0; i < 8; ++i) O[nc][i] = 0.f;
    for (int kk = 0; kk < Ln; kk += 32) {
      v16bf pa = load_a_rm(pb, Ln, kk);
#pragma unroll
      for (int nc = 0; nc < 4; ++nc)
        O[nc] = wmma_bf16(pa, load_b_nm(Vt + (size_t)(nc * 16) * Ln, Ln, kk), O[nc]);
    }

#pragma unroll
    for (int nc = 0; nc < 4; ++nc)
#pragma unroll
      for (int v = 0; v < 8; ++v) {
        int row = r0 + v + 8 * half;
        int col = head * DHn + nc * 16 + (lane & 15);
        ao[obase + (size_t)row * Cn + col] = f32_bf16(O[nc][v] / sm[v]);
      }
  }
}

// ---------------------------------------------------------------------------
// K6: output projection + n_attn-weighted combine + bias mix + ReLU + residual.
// Block = 8 waves, each wave a 32x32 register tile (2x2 wmma per k-step) ->
// block tile 64x128 of (l,c) for one (b,j); loops the 4 slots.
// ---------------------------------------------------------------------------
__global__ __launch_bounds__(256) void k_oproj(
    const unsigned short* __restrict__ ao, const unsigned short* __restrict__ wt,
    const float* __restrict__ bo, const float* __restrict__ n_attn,
    const int* __restrict__ indx, const float* __restrict__ resid,
    float* __restrict__ dstf, unsigned short* __restrict__ dstbf) {
  int bj = blockIdx.z, j = bj & (Hn - 1);
  int w = threadIdx.x >> 5, wr = w >> 2, wc = w & 3;
  int lane = threadIdx.x & 31, n = lane & 15, half = lane >> 4;
  int row0 = blockIdx.y * 64 + wr * 32;                  // l
  int col0 = blockIdx.x * 128 + wc * 32;                 // c

  float tot[2][2][8];
  float bmix[2] = {0.f, 0.f};
#pragma unroll
  for (int i = 0; i < 2; ++i)
#pragma unroll
    for (int jn = 0; jn < 2; ++jn)
#pragma unroll
      for (int v = 0; v < 8; ++v) tot[i][jn][v] = 0.f;

  for (int kslot = 0; kslot < Kt; ++kslot) {
    int idx = indx[bj * Kt + kslot];
    int e = (idx == j) ? 0 : (idx > j ? 2 : 1);
    float wgt = n_attn[bj * Kt + kslot];
    const unsigned short* A  = ao + (((size_t)bj * Kt + kslot) * Ln + row0) * Cn;
    const unsigned short* Bt = wt + (size_t)(3 * 3 + e) * Cn * Cn + (size_t)col0 * Cn;

    v8f acc[2][2];
#pragma unroll
    for (int i = 0; i < 2; ++i)
#pragma unroll
      for (int jn = 0; jn < 2; ++jn)
#pragma unroll
        for (int q = 0; q < 8; ++q) acc[i][jn][q] = 0.f;

    for (int kk = 0; kk < Cn; kk += 32) {
      v16bf a0 = load_a_rm(A,                    Cn, kk);
      v16bf a1 = load_a_rm(A + (size_t)16 * Cn,  Cn, kk);
      v16bf b0 = load_b_nm(Bt,                   Cn, kk);
      v16bf b1 = load_b_nm(Bt + (size_t)16 * Cn, Cn, kk);
      acc[0][0] = wmma_bf16(a0, b0, acc[0][0]);
      acc[0][1] = wmma_bf16(a0, b1, acc[0][1]);
      acc[1][0] = wmma_bf16(a1, b0, acc[1][0]);
      acc[1][1] = wmma_bf16(a1, b1, acc[1][1]);
    }
#pragma unroll
    for (int i = 0; i < 2; ++i)
#pragma unroll
      for (int jn = 0; jn < 2; ++jn)
#pragma unroll
        for (int v = 0; v < 8; ++v) tot[i][jn][v] += wgt * acc[i][jn][v];
#pragma unroll
    for (int jn = 0; jn < 2; ++jn)
      bmix[jn] += wgt * bo[e * Cn + col0 + jn * 16 + n];
  }

#pragma unroll
  for (int i = 0; i < 2; ++i)
#pragma unroll
    for (int jn = 0; jn < 2; ++jn)
#pragma unroll
      for (int v = 0; v < 8; ++v) {
        int row = row0 + i * 16 + v + 8 * half;
        int col = col0 + jn * 16 + n;
        size_t oix = ((size_t)bj * Ln + row) * Cn + col;
        float val = fmaxf(tot[i][jn][v] + bmix[jn], 0.f) + resid[oix];  // relu + resid
        dstf[oix] = val;
        if (dstbf) dstbf[oix] = f32_bf16(val);
      }
}

// ---------------------------------------------------------------------------
extern "C" void kernel_launch(void* const* d_in, const int* in_sizes, int n_in,
                              void* d_out, int out_size, void* d_ws, size_t ws_size,
                              hipStream_t stream) {
  const float* gmap = (const float*)d_in[0];
  const float* hist = (const float*)d_in[1];
  const float* cur  = (const float*)d_in[2];
  // d_in[3..5]: masks (all-ones, unused by the reference math)
  const float* Wq = (const float*)d_in[6];
  const float* bq = (const float*)d_in[7];
  const float* Wk = (const float*)d_in[8];
  const float* bk = (const float*)d_in[9];
  const float* Wv = (const float*)d_in[10];
  const float* bv = (const float*)d_in[11];
  const float* Wo = (const float*)d_in[12];
  const float* bo = (const float*)d_in[13];
  float* out = (float*)d_out;

  char* ws = (char*)d_ws;
  size_t off = 0;
  auto walloc = [&](size_t bytes) -> void* {
    void* p = ws + off;
    off = (off + bytes + 255) & ~(size_t)255;
    return p;
  };
  const size_t nBHLC = (size_t)Bn * Hn * Ln * Cn;        // 2,097,152
  float*          n_attn = (float*)walloc(Bn * Hn * Kt * 4);
  int*            indx   = (int*)walloc(Bn * Hn * Kt * 4);
  unsigned short* wt     = (unsigned short*)walloc((size_t)12 * Cn * Cn * 2);
  float*          ctxf   = (float*)walloc(nBHLC * 4);
  unsigned short* ctxbf  = (unsigned short*)walloc(nBHLC * 2);
  unsigned short* qb     = (unsigned short*)walloc(3 * nBHLC * 2);
  unsigned short* kb     = (unsigned short*)walloc(3 * nBHLC * 2);
  unsigned short* vtb    = (unsigned short*)walloc(3 * nBHLC * 2);
  unsigned short* ao     = (unsigned short*)walloc((size_t)Kt * nBHLC * 2);

  k_topk<<<1, 64, 0, stream>>>(gmap, n_attn, indx);
  k_prepw<<<(3 * Cn * Cn) / 256, 256, 0, stream>>>(Wq, Wk, Wv, Wo, wt);
  k_concat<<<(int)(nBHLC / 256), 256, 0, stream>>>(hist, cur, ctxf, ctxbf);

  for (int layer = 0; layer < 2; ++layer) {
    k_qkv<<<dim3(Cn / 128, (Bn * Hn * Ln) / 64, 9), 256, 0, stream>>>(
        ctxbf, wt, bq, bk, bv, qb, kb, vtb);
    k_attn<<<dim3(NHn, Bn * Hn * Kt), 256, 0, stream>>>(qb, kb, vtb, indx, ao);
    k_oproj<<<dim3(Cn / 128, Ln / 64, Bn * Hn), 256, 0, stream>>>(
        ao, wt, bo, n_attn, indx, ctxf,
        (layer == 0) ? ctxf : out,
        (layer == 0) ? ctxbf : (unsigned short*)nullptr);
  }
}
